// TransICDHead_35596688949772
// MI455X (gfx1250) — matrix operand order repaired and checked by reading, not compiled, not run on hardware
//
#include <hip/hip_runtime.h>

typedef __attribute__((ext_vector_type(16))) _Float16 v16h;
typedef __attribute__((ext_vector_type(8)))  _Float16 v8h;
typedef __attribute__((ext_vector_type(8)))  float    v8f;
typedef __attribute__((ext_vector_type(4)))  float    v4f;

#define B_   8
#define S_   4096
#define H_   768
#define L_   1000
#define LPAD 1024
#define APAD 40   // padded LDS row stride (halves): 32 data + 8 pad (80B, 16B-aligned)

// ---------------- CDNA5 primitives ----------------

__device__ __forceinline__ v8f wmma16(v16h a, v16h b, v8f c) {
  // 8 args: (neg_a, A, neg_b, B, c_mod, C, reuse_a, reuse_b)
  return __builtin_amdgcn_wmma_f32_16x16x32_f16(false, a, false, b, (short)0, c,
                                                false, false);
}

// Async copy 16B global -> LDS (ASYNCcnt-tracked, no VGPR destination).
__device__ __forceinline__ void async_b128(unsigned int lds_off, const void* g) {
  asm volatile("global_load_async_to_lds_b128 %0, %1, off"
               :: "v"(lds_off), "v"((unsigned long long)(uintptr_t)g)
               : "memory");
}

__device__ __forceinline__ void wait_async() {
  asm volatile("s_wait_asynccnt 0" ::: "memory");
}

// ---------------- LDS fragment loaders ----------------
// A-frag (16x32 MxK): lanes 0-15 (M=lane): K 0..7 | 16..23 ; lanes 16-31: K 8..15 | 24..31
__device__ __forceinline__ v16h lds_a_frag(const _Float16* Abase /* row0 */, int lane) {
  int m = lane & 15, sel = lane >> 4;
  const _Float16* p = Abase + m * APAD + sel * 8;
  v8h lo = *(const v8h*)p;
  v8h hi = *(const v8h*)(p + 16);
  v16h a;
#pragma unroll
  for (int i = 0; i < 8; ++i) { a[i] = lo[i]; a[8 + i] = hi[i]; }
  return a;
}

// B-frag (32x16 KxN) stored as [N rows][32 K halves]: lanes 0-15 (N=lane): K 0..7 | 16..23 ; 16-31: K 8..15 | 24..31
__device__ __forceinline__ v16h lds_b_frag(const _Float16* Bbase /* n0 row */, int lane) {
  int n = lane & 15, ksel = lane >> 4;
  const _Float16* p = Bbase + n * APAD + ksel * 8;
  v8h lo = *(const v8h*)p;
  v8h hi = *(const v8h*)(p + 16);
  v16h b;
#pragma unroll
  for (int i = 0; i < 8; ++i) { b[i] = lo[i]; b[8 + i] = hi[i]; }
  return b;
}

// Stage a 128x32 f16 tile (row-major, leading dim ld) into LDS via async copies.
// 512 chunks of 16B; 256 threads handle 2 chunks each.
__device__ __forceinline__ void stage_async_f16(_Float16* lds_tile,
                                                const _Float16* gbase,
                                                int ld, int tid) {
#pragma unroll
  for (int i = 0; i < 2; ++i) {
    int id = tid + i * 256;
    int r = id >> 2, c = id & 3;
    unsigned int lds_off = (unsigned int)(uintptr_t)(lds_tile + r * APAD + c * 8);
    async_b128(lds_off, gbase + (size_t)r * ld + c * 8);
  }
}

// Stage a 128x32 tile from fp32 (row clamp) with cvt->f16, via ds_store.
__device__ __forceinline__ void stage_cvt_f32(_Float16* lds_tile,
                                              const float* gbase, int row0,
                                              int ld, int k0, int tid) {
  int r = tid >> 1, halfsel = tid & 1;          // 128 rows x 2 half-rows of 16
  int row = row0 + r; if (row > L_ - 1) row = L_ - 1;
  const float* p = gbase + (size_t)row * ld + k0 + halfsel * 16;
  v4f f0 = *(const v4f*)p;
  v4f f1 = *(const v4f*)(p + 4);
  v4f f2 = *(const v4f*)(p + 8);
  v4f f3 = *(const v4f*)(p + 12);
  v8h h0v, h1v;
#pragma unroll
  for (int i = 0; i < 4; ++i) {
    h0v[i]     = (_Float16)f0[i];
    h0v[4 + i] = (_Float16)f1[i];
    h1v[i]     = (_Float16)f2[i];
    h1v[4 + i] = (_Float16)f3[i];
  }
  _Float16* d = lds_tile + r * APAD + halfsel * 16;
  *(v8h*)d = h0v;
  *(v8h*)(d + 8) = h1v;
}

// ---------------- Phase 0: precision staging ----------------

__global__ void cvt_tokens_kernel(const float* __restrict__ tok,
                                  _Float16* __restrict__ t16,
                                  _Float16* __restrict__ t16T) {
  size_t idx = (size_t)blockIdx.x * blockDim.x + threadIdx.x;
  if (idx >= (size_t)B_ * S_ * H_) return;
  int h = (int)(idx % H_);
  size_t bs = idx / H_;
  int s = (int)(bs % S_);
  int b = (int)(bs / S_);
  _Float16 hv = (_Float16)tok[idx];
  t16[idx] = hv;                                  // [B,S,H]
  t16T[((size_t)b * H_ + h) * S_ + s] = hv;       // [B,H,S]
}

__global__ void cvt_lq_kernel(const float* __restrict__ lq,
                              _Float16* __restrict__ lq16) {
  int idx = blockIdx.x * blockDim.x + threadIdx.x;
  if (idx >= LPAD * H_) return;
  int row = idx / H_;
  lq16[idx] = (row < L_) ? (_Float16)lq[idx] : (_Float16)0.0f;
}

// ---------------- Phase 1: scores = lq @ tokens^T (per batch) ----------------
// Block tile 128(L) x 128(S), 8 waves, each wave 32x64. K = H = 768.

__global__ void __launch_bounds__(256)
scores_kernel(const _Float16* __restrict__ lq16,
              const _Float16* __restrict__ tok16,
              float* __restrict__ attn_out) {
  __shared__ __align__(128) _Float16 As[2][128][APAD];
  __shared__ __align__(128) _Float16 Bs[2][128][APAD];
  int tid = threadIdx.x, lane = tid & 31, w = tid >> 5;
  int wm = w & 3, wn = w >> 2;                 // 4 M-strips x 2 N-strips
  int s0 = blockIdx.x * 128, l0 = blockIdx.y * 128, b = blockIdx.z;
  const _Float16* tb = tok16 + (size_t)b * S_ * H_;

  v8f acc[2][4] = {};

  stage_async_f16(&As[0][0][0], lq16 + (size_t)l0 * H_, H_, tid);
  stage_async_f16(&Bs[0][0][0], tb + (size_t)s0 * H_, H_, tid);

  int buf = 0;
  for (int k0 = 0; k0 < H_; k0 += 32) {
    wait_async();
    __syncthreads();
    if (k0 + 32 < H_) {
      stage_async_f16(&As[buf ^ 1][0][0], lq16 + (size_t)l0 * H_ + (k0 + 32), H_, tid);
      stage_async_f16(&Bs[buf ^ 1][0][0], tb + (size_t)s0 * H_ + (k0 + 32), H_, tid);
    }
    v16h a0 = lds_a_frag(&As[buf][wm * 32][0], lane);
    v16h a1 = lds_a_frag(&As[buf][wm * 32 + 16][0], lane);
    v16h bf[4];
#pragma unroll
    for (int f = 0; f < 4; ++f) bf[f] = lds_b_frag(&Bs[buf][wn * 64 + f * 16][0], lane);
#pragma unroll
    for (int f = 0; f < 4; ++f) {
      acc[0][f] = wmma16(a0, bf[f], acc[0][f]);
      acc[1][f] = wmma16(a1, bf[f], acc[1][f]);
    }
    buf ^= 1;
  }

  // C/D: lane<16 -> (M=r, N=lane); lane>=16 -> (M=8+r, N=lane-16)
  int n = lane & 15, msel = (lane >> 4) * 8;
  float* outb = attn_out + (size_t)b * L_ * S_;
#pragma unroll
  for (int mi = 0; mi < 2; ++mi) {
#pragma unroll
    for (int r = 0; r < 8; ++r) {
      int l = l0 + wm * 32 + mi * 16 + msel + r;
      if (l < L_) {
        float* orow = outb + (size_t)l * S_ + s0 + wn * 64 + n;
#pragma unroll
        for (int f = 0; f < 4; ++f) orow[f * 16] = acc[mi][f][r];
      }
    }
  }
}

// ---------------- Phase 2: masked softmax in-place over S ----------------

__global__ void softmax_kernel(float* __restrict__ attn,
                               const int* __restrict__ mask) {
  int b = blockIdx.x / L_;
  int l = blockIdx.x % L_;
  float* row = attn + ((size_t)b * L_ + l) * S_;
  const int* mrow = mask + (size_t)b * S_;
  __shared__ float sm[256];

  float lmax = -INFINITY;
  for (int s = threadIdx.x; s < S_; s += 256) {
    float v = row[s];
    if (mrow[s] != 0 && v > lmax) lmax = v;
  }
  sm[threadIdx.x] = lmax;
  __syncthreads();
  for (int off = 128; off > 0; off >>= 1) {
    if (threadIdx.x < off)
      sm[threadIdx.x] = fmaxf(sm[threadIdx.x], sm[threadIdx.x + off]);
    __syncthreads();
  }
  float rmax = sm[0];
  __syncthreads();

  float lsum = 0.0f;
  for (int s = threadIdx.x; s < S_; s += 256) {
    float e = (mrow[s] != 0) ? __expf(row[s] - rmax) : 0.0f;
    row[s] = e;
    lsum += e;
  }
  sm[threadIdx.x] = lsum;
  __syncthreads();
  for (int off = 128; off > 0; off >>= 1) {
    if (threadIdx.x < off) sm[threadIdx.x] += sm[threadIdx.x + off];
    __syncthreads();
  }
  float inv = 1.0f / sm[0];
  for (int s = threadIdx.x; s < S_; s += 256) row[s] *= inv;
}

// ---------------- Phase 3: code_reps = attn @ tokens (per batch) ----------------
// Block tile 128(L) x 128(H), 8 waves, each wave 32x64. K = S = 4096.
// A staged from fp32 attn with cvt; B staged async from transposed f16 tokens.

__global__ void __launch_bounds__(256)
codereps_kernel(const float* __restrict__ attn,
                const _Float16* __restrict__ tok16T,
                float* __restrict__ cr_out) {
  __shared__ __align__(128) _Float16 As[2][128][APAD];
  __shared__ __align__(128) _Float16 Bs[2][128][APAD];
  int tid = threadIdx.x, lane = tid & 31, w = tid >> 5;
  int wm = w & 3, wn = w >> 2;
  int h0 = blockIdx.x * 128, l0 = blockIdx.y * 128, b = blockIdx.z;
  const float* ab = attn + (size_t)b * L_ * S_;
  const _Float16* tb = tok16T + (size_t)b * H_ * S_;

  v8f acc[2][4] = {};

  stage_cvt_f32(&As[0][0][0], ab, l0, S_, 0, tid);
  stage_async_f16(&Bs[0][0][0], tb + (size_t)h0 * S_, S_, tid);

  int buf = 0;
  for (int k0 = 0; k0 < S_; k0 += 32) {
    wait_async();
    __syncthreads();
    if (k0 + 32 < S_) {
      stage_cvt_f32(&As[buf ^ 1][0][0], ab, l0, S_, k0 + 32, tid);
      stage_async_f16(&Bs[buf ^ 1][0][0], tb + (size_t)h0 * S_ + (k0 + 32), S_, tid);
    }
    v16h a0 = lds_a_frag(&As[buf][wm * 32][0], lane);
    v16h a1 = lds_a_frag(&As[buf][wm * 32 + 16][0], lane);
    v16h bf[4];
#pragma unroll
    for (int f = 0; f < 4; ++f) bf[f] = lds_b_frag(&Bs[buf][wn * 64 + f * 16][0], lane);
#pragma unroll
    for (int f = 0; f < 4; ++f) {
      acc[0][f] = wmma16(a0, bf[f], acc[0][f]);
      acc[1][f] = wmma16(a1, bf[f], acc[1][f]);
    }
    buf ^= 1;
  }

  int n = lane & 15, msel = (lane >> 4) * 8;
  float* outb = cr_out + (size_t)b * L_ * H_;
#pragma unroll
  for (int mi = 0; mi < 2; ++mi) {
#pragma unroll
    for (int r = 0; r < 8; ++r) {
      int l = l0 + wm * 32 + mi * 16 + msel + r;
      if (l < L_) {
        float* orow = outb + (size_t)l * H_ + h0 + wn * 64 + n;
#pragma unroll
        for (int f = 0; f < 4; ++f) orow[f * 16] = acc[mi][f][r];
      }
    }
  }
}

// ---------------- Phase 4: logits[b,l] = dot(code_reps[b,l], W[l]) + bias[l] ----------------

__global__ void logits_kernel(const float* __restrict__ cr,
                              const float* __restrict__ W,
                              const float* __restrict__ bias,
                              float* __restrict__ logits) {
  int gw = blockIdx.x * (blockDim.x >> 5) + (threadIdx.x >> 5);
  int lane = threadIdx.x & 31;
  if (gw >= B_ * L_) return;
  int l = gw % L_;
  const float* x = cr + (size_t)gw * H_;
  const float* w = W + (size_t)l * H_;
  float acc = 0.0f;
  for (int h = lane; h < H_; h += 32) acc = fmaf(x[h], w[h], acc);
#pragma unroll
  for (int off = 16; off > 0; off >>= 1) acc += __shfl_down(acc, off, 32);
  if (lane == 0) logits[gw] = acc + bias[l];
}

// ---------------- Launch ----------------

extern "C" void kernel_launch(void* const* d_in, const int* in_sizes, int n_in,
                              void* d_out, int out_size, void* d_ws, size_t ws_size,
                              hipStream_t stream) {
  const float* tokens = (const float*)d_in[0];
  const int*   mask   = (const int*)d_in[1];
  const float* lq     = (const float*)d_in[2];
  const float* W      = (const float*)d_in[3];
  const float* bias   = (const float*)d_in[4];

  float* out    = (float*)d_out;
  float* logits = out;                                // [B,L]
  float* attn   = out + (size_t)B_ * L_;              // [B,L,S]
  float* creps  = attn + (size_t)B_ * L_ * S_;        // [B,L,H]

  char* ws = (char*)d_ws;
  _Float16* tok16  = (_Float16*)ws;                               // [B,S,H] f16
  _Float16* tok16T = (_Float16*)(ws + (size_t)B_ * S_ * H_ * 2);  // [B,H,S] f16
  _Float16* lq16   = (_Float16*)(ws + (size_t)B_ * S_ * H_ * 4);  // [LPAD,H] f16

  {
    size_t n = (size_t)B_ * S_ * H_;
    cvt_tokens_kernel<<<(unsigned)((n + 255) / 256), 256, 0, stream>>>(tokens, tok16, tok16T);
    int n2 = LPAD * H_;
    cvt_lq_kernel<<<(n2 + 255) / 256, 256, 0, stream>>>(lq, lq16);
  }

  scores_kernel<<<dim3(S_ / 128, LPAD / 128, B_), 256, 0, stream>>>(lq16, tok16, attn);
  softmax_kernel<<<B_ * L_, 256, 0, stream>>>(attn, mask);
  codereps_kernel<<<dim3(H_ / 128, LPAD / 128, B_), 256, 0, stream>>>(attn, tok16T, creps);
  logits_kernel<<<(B_ * L_ + 7) / 8, 256, 0, stream>>>(creps, W, bias, logits);
}
